// ResidualBlock_49572512530868
// MI455X (gfx1250) — matrix operand order repaired and checked by reading, not compiled, not run on hardware
//
#include <hip/hip_runtime.h>
#include <hip/hip_bf16.h>

// ---------------------------------------------------------------------------
// Mamba residual block for MI455X (gfx1250, wave32, WMMA).
// All four GEMMs run on v_wmma_f32_16x16x32_f16; scan is lane-parallel.
// ---------------------------------------------------------------------------

#define B_SZ     2
#define L_SEQ    2048
#define D_MODEL  768
#define D_INNER  1536
#define D_STATE  16
#define DT_RANK  48
#define DT_PAD   64          // DT_RANK padded to a multiple of 32 for WMMA K
#define D_CONV   4
#define NTOK     (B_SZ * L_SEQ)   // 4096 tokens

typedef __attribute__((ext_vector_type(16))) _Float16 v16h;
typedef __attribute__((ext_vector_type(8)))  _Float16 v8h;
typedef __attribute__((ext_vector_type(8)))  float    v8f;

// ---------------------------------------------------------------------------
// WMMA fragment loads (wave32; contiguous K order k = 16*(lane/16) + s in A and B)
// ---------------------------------------------------------------------------

// A fragment: 16x32 f16 tile of row-major A[M,K].
// lane L: row = L&15; K-halves: lanes 0-15 -> K 0..15, lanes 16-31 -> K 16..31
__device__ __forceinline__ v16h load_a_frag(const _Float16* __restrict__ A,
                                            int lda, int m0, int k0) {
  const int lane = threadIdx.x & 31;
  const int m    = lane & 15;
  const int koff = (lane >> 4) << 4;                 // 0 or 16
  const _Float16* p = A + (size_t)(m0 + m) * lda + (k0 + koff);
  union { v16h v; v8h h[2]; } u;
  u.h[0] = *(const v8h*)(p);                         // K = koff .. koff+7
  u.h[1] = *(const v8h*)(p + 8);                     // K = koff+8 .. koff+15
  return u.v;
}

// B fragment: 32x16 f16 tile of B = W^T where W is row-major [N,K].
// lane L: col n = L&15; lanes 0-15 hold K=0..15, lanes 16-31 hold K=16..31.
// Contiguous 16 f16 along K of W's row -> single 32-byte load.
__device__ __forceinline__ v16h load_b_frag(const _Float16* __restrict__ W,
                                            int ldw, int n0, int k0) {
  const int lane = threadIdx.x & 31;
  const int n    = lane & 15;
  const int koff = (lane >> 4) << 4;                 // 0 or 16
  return *(const v16h*)(W + (size_t)(n0 + n) * ldw + (k0 + koff));
}

// ---------------------------------------------------------------------------
// Generic WMMA GEMM: C[M,N] = A[M,K] * W[N,K]^T, f16 in, f32 out.
// One wave per 16 x (16*NSUB) output tile: A-fragment reused NSUB times.
// EPI: 0=store  1=x_proj(dual-store dt cols to padded f16)
//      2=+bias,softplus  3=+residual
// ---------------------------------------------------------------------------
template <int NSUB, int EPI>
__global__ __launch_bounds__(32)
void mamba_gemm(const _Float16* __restrict__ A, const _Float16* __restrict__ W,
                float* __restrict__ C, int M, int N, int K,
                const float* __restrict__ bias, const float* __restrict__ resid,
                _Float16* __restrict__ aux) {
  const int lane = threadIdx.x;
  const int m0 = blockIdx.y * 16;
  const int n0 = blockIdx.x * (16 * NSUB);

  v8f acc[NSUB];
#pragma unroll
  for (int s = 0; s < NSUB; ++s) acc[s] = (v8f){};

  for (int k0 = 0; k0 < K; k0 += 32) {
    v16h a = load_a_frag(A, K, m0, k0);
#pragma unroll
    for (int s = 0; s < NSUB; ++s) {
      v16h b = load_b_frag(W, K, n0 + 16 * s, k0);
      acc[s] = __builtin_amdgcn_wmma_f32_16x16x32_f16(
          /*neg_a=*/false, a, /*neg_b=*/false, b,
          /*c_mod=*/(short)0, acc[s], /*reuse_a=*/false, /*reuse_b=*/false);
    }
  }

  // C/D layout: lane -> col = lane&15, rows mbase..mbase+7, mbase = (lane>>4)*8
  const int n  = lane & 15;
  const int mb = (lane >> 4) << 3;
#pragma unroll
  for (int s = 0; s < NSUB; ++s) {
    const int col = n0 + 16 * s + n;
#pragma unroll
    for (int v = 0; v < 8; ++v) {
      const int row = m0 + mb + v;
      float val = acc[s][v];
      if (EPI == 1) {                       // x_proj: also stash dt cols as f16
        C[(size_t)row * N + col] = val;
        if (col < DT_RANK) aux[(size_t)row * DT_PAD + col] = (_Float16)val;
      } else if (EPI == 2) {                // dt_proj: bias + softplus
        float t = val + bias[col];
        C[(size_t)row * N + col] = (t > 20.f) ? t : log1pf(__expf(t));
      } else if (EPI == 3) {                // out_proj: + residual
        C[(size_t)row * N + col] = val + resid[(size_t)row * N + col];
      } else {
        C[(size_t)row * N + col] = val;
      }
    }
  }
}

// ---------------------------------------------------------------------------
// RMSNorm -> f16 activations for in_proj GEMM
// ---------------------------------------------------------------------------
__global__ __launch_bounds__(256)
void mamba_rmsnorm(const float* __restrict__ x, const float* __restrict__ w,
                   _Float16* __restrict__ out) {
  const int t = blockIdx.x;
  const float* xr = x + (size_t)t * D_MODEL;
  float s = 0.f;
  for (int i = threadIdx.x; i < D_MODEL; i += 256) { float v = xr[i]; s += v * v; }
  __shared__ float red[256];
  red[threadIdx.x] = s;
  __syncthreads();
  for (int off = 128; off > 0; off >>= 1) {
    if (threadIdx.x < off) red[threadIdx.x] += red[threadIdx.x + off];
    __syncthreads();
  }
  const float rs = rsqrtf(red[0] / (float)D_MODEL + 1e-5f);
  for (int i = threadIdx.x; i < D_MODEL; i += 256)
    out[(size_t)t * D_MODEL + i] = (_Float16)(xr[i] * rs * w[i]);
}

// ---------------------------------------------------------------------------
// Causal depthwise conv (K=4) + SiLU.  xz layout [NTOK, 2*D_INNER]; xh = cols 0..Di-1
// ---------------------------------------------------------------------------
__global__ __launch_bounds__(256)
void mamba_conv_silu(const float* __restrict__ xz, const float* __restrict__ cw,
                     const float* __restrict__ cb, float* __restrict__ xc,
                     _Float16* __restrict__ xc16) {
  const int idx = blockIdx.x * blockDim.x + threadIdx.x;   // over NTOK*D_INNER
  const int d = idx % D_INNER;
  const int t = idx / D_INNER;
  const int l = t % L_SEQ;
  float s = cb[d];
#pragma unroll
  for (int j = 0; j < D_CONV; ++j) {
    const int ll = l - (D_CONV - 1) + j;
    if (ll >= 0)
      s += cw[d * D_CONV + j] * xz[(size_t)(t - (D_CONV - 1) + j) * (2 * D_INNER) + d];
  }
  s = s / (1.f + __expf(-s));                              // SiLU
  xc[idx]   = s;
  xc16[idx] = (_Float16)s;
}

// ---------------------------------------------------------------------------
// Selective scan. 16 lanes per (b,d) channel: lane n owns state n.
// y = sum_n h[n]*C[n] via shfl_xor tree within 16-lane groups.
// Then add D skip, gate with silu(z), emit f16 for out_proj GEMM.
// ---------------------------------------------------------------------------
__global__ __launch_bounds__(256)
void mamba_scan(const float* __restrict__ delta, const float* __restrict__ xc,
                const float* __restrict__ proj, const float* __restrict__ A_log,
                const float* __restrict__ Dp, const float* __restrict__ xz,
                _Float16* __restrict__ yg16) {
  const int gid = blockIdx.x * blockDim.x + threadIdx.x;   // NTOK-independent: B*Di*16
  const int n  = gid & 15;
  const int ch = gid >> 4;                                  // 0 .. B*Di-1
  const int d  = ch % D_INNER;
  const int b  = ch / D_INNER;
  const float A  = -__expf(A_log[d * D_STATE + n]);
  const float Dv = Dp[d];
  float h = 0.f;
  for (int l = 0; l < L_SEQ; ++l) {
    const size_t t  = (size_t)b * L_SEQ + l;
    const float dl  = delta[t * D_INNER + d];
    const float xv  = xc[t * D_INNER + d];
    const float Bm  = proj[t * (DT_RANK + 2 * D_STATE) + DT_RANK + n];
    const float Cm  = proj[t * (DT_RANK + 2 * D_STATE) + DT_RANK + D_STATE + n];
    h = __expf(dl * A) * h + dl * Bm * xv;
    float y = h * Cm;
    y += __shfl_xor(y, 1, 16);
    y += __shfl_xor(y, 2, 16);
    y += __shfl_xor(y, 4, 16);
    y += __shfl_xor(y, 8, 16);
    if (n == 0) {
      y += xv * Dv;                                        // D skip
      const float z = xz[t * (2 * D_INNER) + D_INNER + d];
      y *= z / (1.f + __expf(-z));                         // gate: y * silu(z)
      yg16[t * D_INNER + d] = (_Float16)y;
    }
  }
}

// ---------------------------------------------------------------------------
// Weight / dtype prep
// ---------------------------------------------------------------------------
__global__ __launch_bounds__(256)
void f32_to_f16_k(const float* __restrict__ in, _Float16* __restrict__ out, int n) {
  const int i = blockIdx.x * blockDim.x + threadIdx.x;
  if (i < n) out[i] = (_Float16)in[i];
}

// dt_proj_w [Di,48] -> f16 [Di,64] zero padded (K multiple of 32 for WMMA)
__global__ __launch_bounds__(256)
void dtw_pad_k(const float* __restrict__ in, _Float16* __restrict__ out) {
  const int i = blockIdx.x * blockDim.x + threadIdx.x;     // Di*64
  const int r = i & (DT_PAD - 1);
  const int e = i >> 6;
  out[i] = (r < DT_RANK) ? (_Float16)in[e * DT_RANK + r] : (_Float16)0.f;
}

// ---------------------------------------------------------------------------
// Launch
// ---------------------------------------------------------------------------
static inline size_t align_up(size_t v, size_t a) { return (v + a - 1) & ~(a - 1); }

extern "C" void kernel_launch(void* const* d_in, const int* in_sizes, int n_in,
                              void* d_out, int out_size, void* d_ws, size_t ws_size,
                              hipStream_t stream) {
  (void)in_sizes; (void)n_in; (void)out_size; (void)ws_size;

  const float* x        = (const float*)d_in[0];
  const float* norm_w   = (const float*)d_in[1];
  const float* in_w     = (const float*)d_in[2];
  const float* conv_w   = (const float*)d_in[3];
  const float* conv_b   = (const float*)d_in[4];
  const float* xproj_w  = (const float*)d_in[5];
  const float* dtproj_w = (const float*)d_in[6];
  const float* dtproj_b = (const float*)d_in[7];
  const float* A_log    = (const float*)d_in[8];
  const float* Dp       = (const float*)d_in[9];
  const float* out_w    = (const float*)d_in[10];
  float* out            = (float*)d_out;

  // ---- workspace carve (256B aligned) ----
  char* ws = (char*)d_ws;
  size_t off = 0;
  auto carve = [&](size_t bytes) { void* p = ws + off; off = align_up(off + bytes, 256); return p; };

  _Float16* inW16  = (_Float16*)carve((size_t)(2 * D_INNER) * D_MODEL * 2);
  _Float16* xW16   = (_Float16*)carve((size_t)(DT_RANK + 2 * D_STATE) * D_INNER * 2);
  _Float16* dtW16  = (_Float16*)carve((size_t)D_INNER * DT_PAD * 2);
  _Float16* outW16 = (_Float16*)carve((size_t)D_MODEL * D_INNER * 2);
  _Float16* h16    = (_Float16*)carve((size_t)NTOK * D_MODEL * 2);
  float*    xz     = (float*)   carve((size_t)NTOK * 2 * D_INNER * 4);
  float*    xc     = (float*)   carve((size_t)NTOK * D_INNER * 4);
  _Float16* xc16   = (_Float16*)carve((size_t)NTOK * D_INNER * 2);
  float*    proj   = (float*)   carve((size_t)NTOK * (DT_RANK + 2 * D_STATE) * 4);
  _Float16* dt16   = (_Float16*)carve((size_t)NTOK * DT_PAD * 2);
  float*    delta  = (float*)   carve((size_t)NTOK * D_INNER * 4);
  _Float16* yg16   = (_Float16*)carve((size_t)NTOK * D_INNER * 2);

  // ---- 1. weights -> f16 (once per call; deterministic) ----
  {
    int n;
    n = 2 * D_INNER * D_MODEL;
    f32_to_f16_k<<<(n + 255) / 256, 256, 0, stream>>>(in_w, inW16, n);
    n = (DT_RANK + 2 * D_STATE) * D_INNER;
    f32_to_f16_k<<<(n + 255) / 256, 256, 0, stream>>>(xproj_w, xW16, n);
    n = D_MODEL * D_INNER;
    f32_to_f16_k<<<(n + 255) / 256, 256, 0, stream>>>(out_w, outW16, n);
    n = D_INNER * DT_PAD;
    dtw_pad_k<<<(n + 255) / 256, 256, 0, stream>>>(dtproj_w, dtW16);
  }

  // ---- 2. RMSNorm -> h16 ----
  mamba_rmsnorm<<<NTOK, 256, 0, stream>>>(x, norm_w, h16);

  // ---- 3. in_proj: xz[NTOK, 3072] = h16 @ in_w^T (WMMA) ----
  mamba_gemm<4, 0><<<dim3((2 * D_INNER) / 64, NTOK / 16), 32, 0, stream>>>(
      h16, inW16, xz, NTOK, 2 * D_INNER, D_MODEL, nullptr, nullptr, nullptr);

  // ---- 4. causal conv + SiLU -> xc (f32 + f16) ----
  mamba_conv_silu<<<(NTOK * D_INNER) / 256, 256, 0, stream>>>(xz, conv_w, conv_b, xc, xc16);

  // ---- 5. x_proj: proj[NTOK, 80] = xc16 @ xproj_w^T; dt cols -> dt16 (padded) ----
  hipMemsetAsync(dt16, 0, (size_t)NTOK * DT_PAD * 2, stream);
  mamba_gemm<1, 1><<<dim3((DT_RANK + 2 * D_STATE) / 16, NTOK / 16), 32, 0, stream>>>(
      xc16, xW16, proj, NTOK, DT_RANK + 2 * D_STATE, D_INNER, nullptr, nullptr, dt16);

  // ---- 6. dt_proj + softplus: delta[NTOK, Di] ----
  mamba_gemm<4, 2><<<dim3(D_INNER / 64, NTOK / 16), 32, 0, stream>>>(
      dt16, dtW16, delta, NTOK, D_INNER, DT_PAD, dtproj_b, nullptr, nullptr);

  // ---- 7. selective scan + D-skip + gating -> yg16 ----
  mamba_scan<<<(B_SZ * D_INNER * D_STATE) / 256, 256, 0, stream>>>(
      delta, xc, proj, A_log, Dp, xz, yg16);

  // ---- 8. out_proj + residual -> d_out[0 : NTOK*D_MODEL] ----
  mamba_gemm<4, 3><<<dim3(D_MODEL / 64, NTOK / 16), 32, 0, stream>>>(
      yg16, outW16, out, NTOK, D_MODEL, D_INNER, nullptr, x, nullptr);

  // ---- 9. tuple's second output: the residual itself ----
  hipMemcpyAsync(out + (size_t)NTOK * D_MODEL, x,
                 (size_t)NTOK * D_MODEL * sizeof(float),
                 hipMemcpyDeviceToDevice, stream);
}
